// RadarSparseFiltBackbone_42606075576914
// MI455X (gfx1250) — matrix operand order repaired and checked by reading, not compiled, not run on hardware
//
#include <hip/hip_runtime.h>
#include <hip/hip_bf16.h>
#include <math.h>

typedef _Float16 f16;
typedef __attribute__((ext_vector_type(16))) _Float16 v16h;
typedef __attribute__((ext_vector_type(8)))  float    v8f;

// ---------------------------------------------------------------------------
// Unconditional fragment loader: two 16-byte chunks -> one v16h (32B).
// Boundary handling = caller selects a zero-padded pointer (no EXEC churn).
// ---------------------------------------------------------------------------
__device__ __forceinline__ v16h frag_load(const f16* p0, const f16* p1) {
  union { uint4 q[2]; v16h h; } u;
  u.q[0] = *reinterpret_cast<const uint4*>(p0);
  u.q[1] = *reinterpret_cast<const uint4*>(p1);
  return u.h;
}

// ---------------------------------------------------------------------------
// CDNA5 async global->LDS staging (ASYNCcnt path).
// 256 threads cooperatively stage 4*KC*1024 bytes (KC in {2,4,8});
// each thread issues exactly KC b128 async loads -> uniform per-wave count.
// ---------------------------------------------------------------------------
__device__ __forceinline__ void stage_async4(const f16* g, unsigned ldsBase, int KC) {
  const int chunk = KC << 4;                                  // bytes per thread
  unsigned ld = ldsBase + threadIdx.x * chunk;
  unsigned long long ga =
      (unsigned long long)(const void*)((const char*)g + threadIdx.x * chunk);
  for (int i = 0; i < KC; ++i) {
    asm volatile("global_load_async_to_lds_b128 %0, %1, off" :: "v"(ld), "v"(ga) : "memory");
    ld += 16; ga += 16;
  }
}

__device__ __forceinline__ void wait_async_kc(int pending) {
  if (pending == 0)      asm volatile("s_wait_asynccnt 0x0" ::: "memory");
  else if (pending == 2) asm volatile("s_wait_asynccnt 0x2" ::: "memory");
  else if (pending == 4) asm volatile("s_wait_asynccnt 0x4" ::: "memory");
  else                   asm volatile("s_wait_asynccnt 0x8" ::: "memory");
}

// ---------------------------------------------------------------------------
// Elementwise / setup kernels
// ---------------------------------------------------------------------------
__global__ void zero_f32(float* __restrict__ p, long n) {
  long i = (long)blockIdx.x * blockDim.x + threadIdx.x;
  long stride = (long)gridDim.x * blockDim.x;
  for (; i < n; i += stride) p[i] = 0.f;
}

__global__ void scatter_pts(const float* __restrict__ feat, const int* __restrict__ idx,
                            float* __restrict__ dense, float* __restrict__ cnt,
                            int Npts, int Z, int Y, int X) {
  int i = blockIdx.x * blockDim.x + threadIdx.x;
  if (i >= Npts) return;
  int b = idx[4 * i + 0], z = idx[4 * i + 1], y = idx[4 * i + 2], x = idx[4 * i + 3];
  size_t v = (((size_t)b * Z + z) * Y + y) * X + x;
  atomicAdd(&cnt[v], 1.f);
  #pragma unroll
  for (int c = 0; c < 4; ++c) atomicAdd(&dense[v * 4 + c], feat[4 * i + c]);
}

__global__ void conv_in_k(const float* __restrict__ dense, const float* __restrict__ cnt,
                          const float* __restrict__ w_in, const float* __restrict__ b_in,
                          f16* __restrict__ out, long NV, int Cout) {
  long i = (long)blockIdx.x * blockDim.x + threadIdx.x;
  if (i >= NV * Cout) return;
  int  c = (int)(i % Cout);
  long v = i / Cout;
  float yv = 0.f;
  if (cnt[v] > 0.f) {
    const float* d = dense + v * 4;
    const float* w = w_in + (size_t)c * 4;
    yv = w[0] * d[0] + w[1] * d[1] + w[2] * d[2] + w[3] * d[3] + b_in[c];
  }
  out[i] = (f16)yv;
}

__global__ void dilate_k(const float* __restrict__ min_, float* __restrict__ mout,
                         int B, int Zi, int Yi, int Xi,
                         int Zo, int Yo, int Xo, int stride) {
  long NV = (long)B * Zo * Yo * Xo;
  long i = (long)blockIdx.x * blockDim.x + threadIdx.x;
  if (i >= NV) return;
  int x = (int)(i % Xo); long t = i / Xo;
  int y = (int)(t % Yo); t /= Yo;
  int z = (int)(t % Zo); int b = (int)(t / Zo);
  float acc = 0.f;
  for (int dz = 0; dz < 3; ++dz) {
    int zi = z * stride + dz - 1;
    if ((unsigned)zi >= (unsigned)Zi) continue;
    for (int dy = 0; dy < 3; ++dy) {
      int yi = y * stride + dy - 1;
      if ((unsigned)yi >= (unsigned)Yi) continue;
      for (int dx = 0; dx < 3; ++dx) {
        int xi = x * stride + dx - 1;
        if ((unsigned)xi >= (unsigned)Xi) continue;
        if (min_[(((size_t)b * Zi + zi) * Yi + yi) * Xi + xi] > 0.f) acc = 1.f;
      }
    }
  }
  mout[i] = acc;
}

__global__ void bn_prep(const float* __restrict__ g, const float* __restrict__ beta,
                        const float* __restrict__ m, const float* __restrict__ v,
                        const float* __restrict__ bconv,
                        float* __restrict__ sc, float* __restrict__ sh, int C) {
  int i = blockIdx.x * blockDim.x + threadIdx.x;
  if (i >= C) return;
  float s = g[i] * rsqrtf(v[i] + 1e-5f);
  sc[i] = s;
  sh[i] = (bconv[i] - m[i]) * s + beta[i];
}

__global__ void f32_to_f16_k(const float* __restrict__ in, f16* __restrict__ out, long n) {
  long i = (long)blockIdx.x * blockDim.x + threadIdx.x;
  if (i < n) out[i] = (f16)in[i];
}

// ---------------------------------------------------------------------------
// Weight pre-packing into this file's WMMA B-matrix (32x16 f16) lane order (contiguous k, matched by A loads):
//   lanes 0-15 hold K=0..15, lanes 16-31 K=16..31; N = lane&15; 16 f16/lane.
// Conv layout: [ngroup][tap][sub(4)][kc] so each (group,tap) panel is one
// contiguous 4*KC KB async stage.
// ---------------------------------------------------------------------------
__global__ void pack_conv_w(const float* __restrict__ w, f16* __restrict__ wp,
                            int Co, int Cin) {
  int  KC = Cin >> 5;
  long total = (long)(Co >> 6) * 27 * 4 * KC * 512;
  long i = (long)blockIdx.x * blockDim.x + threadIdx.x;
  if (i >= total) return;
  int e = (int)(i & 15); long t = i >> 4;
  int lane = (int)(t & 31); t >>= 5;
  int kc = (int)(t % KC); t /= KC;
  int sub = (int)(t & 3); t >>= 2;
  int tap = (int)(t % 27); int g = (int)(t / 27);
  int co = (((g << 2) + sub) << 4) + (lane & 15);
  int ci = (kc << 5) + e + ((lane >> 4) << 4);
  wp[i] = (f16)w[((size_t)co * Cin + ci) * 27 + tap];   // OIDHW, tap=(dz*3+dy)*3+dx
}

// Linear layout: [nt][kc] -- a 4-nt group's panel is already contiguous.
__global__ void pack_lin_w(const float* __restrict__ w, f16* __restrict__ wp,
                           int Nout, int K) {
  int  KC = K >> 5;
  long total = (long)(Nout >> 4) * KC * 512;
  long i = (long)blockIdx.x * blockDim.x + threadIdx.x;
  if (i >= total) return;
  int e = (int)(i & 15); long t = i >> 4;
  int lane = (int)(t & 31); t >>= 5;
  int kc = (int)(t % KC); int nt = (int)(t / KC);
  int n = (nt << 4) + (lane & 15);
  int k = (kc << 5) + e + ((lane >> 4) << 4);
  wp[i] = (f16)w[(size_t)n * K + k];                    // y = x @ W.T => B(k,n)=W[n,k]
}

// ---------------------------------------------------------------------------
// Implicit-GEMM 3x3x3 conv, f16 NDHWC in/out, f32 accumulate via WMMA.
// Block = 8 waves sharing one 64-wide output-channel group; per tap the
// 4*KC KB weight panel is async-staged into LDS (double-buffered over taps).
// Each wave computes a 16-voxel x 64-channel tile: one A fragment feeds 4
// WMMAs (4x arithmetic intensity on the activation stream).
// Epilogue: fused BN scale/shift + ReLU + occupancy mask.
// ---------------------------------------------------------------------------
__global__ __launch_bounds__(256) void conv3d_wmma(
    const f16* __restrict__ in, f16* __restrict__ out,
    const f16* __restrict__ wpack, const f16* __restrict__ zpad,
    const float* __restrict__ sc, const float* __restrict__ sh,
    const float* __restrict__ mask,
    int B, int Zi, int Yi, int Xi, int Cin,
    int Zo, int Yo, int Xo, int Co, int stride) {
  extern __shared__ f16 smem[];                 // 2 * 4*KC*512 f16 (double buffer)
  const int nG  = Co >> 6;                      // 64-channel groups
  const int ntX = (Xo + 15) >> 4;
  const long S  = (long)B * Zo * Yo * ntX;      // spatial tiles
  const int wv   = threadIdx.x >> 5;
  const int lane = threadIdx.x & 31;
  const int mlane = lane & 15, khalf = lane >> 4;

  const int  g    = (int)(blockIdx.x % nG);
  const long sid  = (long)(blockIdx.x / nG) * 8 + wv;
  const bool sval = sid < S;
  const long sidc = sval ? sid : (S - 1);

  int xt = (int)(sidc % ntX); long t = sidc / ntX;
  int yo = (int)(t % Yo); t /= Yo;
  int zo = (int)(t % Zo); int b = (int)(t / Zo);

  const int KC = Cin >> 5;
  const unsigned lds0 = (unsigned)(size_t)&smem[0];
  const unsigned panelB = (unsigned)(KC << 12);           // 4*KC*1024 bytes

  const f16* wg = wpack + ((size_t)g * 27 * 4 * KC << 9); // 512 f16 per block

  const int xo_l = (xt << 4) + mlane;
  const bool xvalid = xo_l < Xo;

  stage_async4(wg, lds0, KC);                   // tap 0 -> buffer 0

  v8f acc[4];
  #pragma unroll
  for (int s2 = 0; s2 < 4; ++s2) acc[s2] = (v8f){0.f,0.f,0.f,0.f,0.f,0.f,0.f,0.f};

  int bufsel = 0;
  for (int tap = 0; tap < 27; ++tap) {
    const bool more = (tap + 1) < 27;
    if (more)
      stage_async4(wg + ((size_t)(tap + 1) * 4 * KC << 9), lds0 + (bufsel ^ 1) * panelB, KC);
    wait_async_kc(more ? KC : 0);               // this tap's panel is in LDS
    __syncthreads();

    const int dz = tap / 9, dy = (tap / 3) % 3, dx = tap % 3;
    const int zi = zo * stride + dz - 1;
    const int yi = yo * stride + dy - 1;
    const int xi = xo_l * stride + dx - 1;
    const bool aok = sval && xvalid &&
                     ((unsigned)zi < (unsigned)Zi) &&
                     ((unsigned)yi < (unsigned)Yi) &&
                     ((unsigned)xi < (unsigned)Xi);
    const f16* arow = aok
        ? in + ((((size_t)b * Zi + zi) * Yi + yi) * Xi + xi) * (size_t)Cin
        : zpad;
    const f16* lbuf = smem + (size_t)bufsel * (KC << 11); // 4*KC*512 f16 per panel
    for (int kc = 0; kc < KC; ++kc) {
      const f16* ap = arow + (kc << 5) + (khalf << 4);
      v16h a = frag_load(ap, ap + 8);
      #pragma unroll
      for (int sub = 0; sub < 4; ++sub) {
        const f16* bp = lbuf + ((sub * KC + kc) << 9) + (lane << 4);
        v16h bf = frag_load(bp, bp + 8);
        acc[sub] = __builtin_amdgcn_wmma_f32_16x16x32_f16(false, a, false, bf,
                                                          (short)0, acc[sub], false, false);
      }
    }
    __syncthreads();                            // buffer reuse fence
    bufsel ^= 1;
  }

  // D layout: lane holds channel N; VGPR r -> voxel M = r + 8*khalf
  int   nn[4];
  float scn[4], shn[4];
  #pragma unroll
  for (int sub = 0; sub < 4; ++sub) {
    nn[sub]  = (((g << 2) + sub) << 4) + mlane;
    scn[sub] = sc[nn[sub]];
    shn[sub] = sh[nn[sub]];
  }
  const size_t obase = (((size_t)b * Zo + zo) * Yo + yo) * (size_t)Xo;
  #pragma unroll
  for (int r = 0; r < 8; ++r) {
    int xo_e = (xt << 4) + r + (khalf << 3);
    if (sval && xo_e < Xo) {
      size_t v = obase + xo_e;
      float mv = mask[v];
      #pragma unroll
      for (int sub = 0; sub < 4; ++sub) {
        float yv = acc[sub][r] * scn[sub] + shn[sub];
        yv = fmaxf(yv, 0.f) * mv;
        out[v * (size_t)Co + nn[sub]] = (f16)yv;
      }
    }
  }
}

// ---------------------------------------------------------------------------
// Generic WMMA GEMM: Y[M,N] = A[M,K](f16) @ Wpack + bias.
// Block = 8 waves sharing one 64-wide N group; K-panel async-staged once.
// Each wave: 16 rows x 64 cols (4 accumulators, A reused 4x).
// emode 0: f16 row-major out.  emode 1: f32 BEV pixel-shuffle scatter.
// ---------------------------------------------------------------------------
__global__ __launch_bounds__(256) void gemm_wmma(
    const f16* __restrict__ A, const f16* __restrict__ wpack,
    const float* __restrict__ bias, const f16* __restrict__ zpad,
    int M, int K, int N,
    int emode, f16* __restrict__ outH, float* __restrict__ outBev,
    int yy, int xx, int r, int choff, int OUTC, int OY, int OX) {
  extern __shared__ f16 smem[];                 // 4*KC*512 f16
  const int nG = N >> 6;
  const int mT = (M + 15) >> 4;
  const int wv   = threadIdx.x >> 5;
  const int lane = threadIdx.x & 31;
  const int mlane = lane & 15, khalf = lane >> 4;

  const int  g    = (int)(blockIdx.x % nG);
  const long mt   = (long)(blockIdx.x / nG) * 8 + wv;
  const bool mval = mt < mT;
  const long mtc  = mval ? mt : (mT - 1);

  const int KC = K >> 5;
  const unsigned lds0 = (unsigned)(size_t)&smem[0];

  stage_async4(wpack + ((size_t)(g << 2) * KC << 9), lds0, KC);
  wait_async_kc(0);
  __syncthreads();

  const int  ma  = (int)(mtc << 4) + mlane;
  const bool aok = mval && (ma < M);
  const f16* arow = aok ? (A + (size_t)ma * K) : zpad;

  v8f acc[4];
  #pragma unroll
  for (int s2 = 0; s2 < 4; ++s2) acc[s2] = (v8f){0.f,0.f,0.f,0.f,0.f,0.f,0.f,0.f};

  for (int kc = 0; kc < KC; ++kc) {
    const f16* ap = arow + (kc << 5) + (khalf << 4);
    v16h a = frag_load(ap, ap + 8);
    #pragma unroll
    for (int sub = 0; sub < 4; ++sub) {
      const f16* bp = smem + ((sub * KC + kc) << 9) + (lane << 4);
      v16h bf = frag_load(bp, bp + 8);
      acc[sub] = __builtin_amdgcn_wmma_f32_16x16x32_f16(false, a, false, bf,
                                                        (short)0, acc[sub], false, false);
    }
  }

  int   nn[4];
  float bb[4];
  #pragma unroll
  for (int sub = 0; sub < 4; ++sub) {
    nn[sub] = (((g << 2) + sub) << 4) + mlane;
    bb[sub] = bias ? bias[nn[sub]] : 0.f;
  }

  if (emode == 0) {
    #pragma unroll
    for (int rr = 0; rr < 8; ++rr) {
      int m = (int)(mtc << 4) + rr + (khalf << 3);
      if (mval && m < M) {
        #pragma unroll
        for (int sub = 0; sub < 4; ++sub)
          outH[(size_t)m * N + nn[sub]] = (f16)(acc[sub][rr] + bb[sub]);
      }
    }
  } else {
    const int rr2 = r * r;
    int ccI[4], ryA[4], rxA[4];
    #pragma unroll
    for (int sub = 0; sub < 4; ++sub) {
      ccI[sub] = nn[sub] / rr2;
      int rem  = nn[sub] % rr2;
      ryA[sub] = rem / r; rxA[sub] = rem % r;
    }
    #pragma unroll
    for (int rr = 0; rr < 8; ++rr) {
      int m = (int)(mtc << 4) + rr + (khalf << 3);
      if (mval && m < M) {
        int x = m % xx; int t2 = m / xx;
        int y = t2 % yy; int b = t2 / yy;
        #pragma unroll
        for (int sub = 0; sub < 4; ++sub) {
          size_t o = (((size_t)b * OUTC + choff + ccI[sub]) * OY + (size_t)y * r + ryA[sub]) * OX
                   + (size_t)x * r + rxA[sub];
          outBev[o] = acc[sub][rr] + bb[sub];
        }
      }
    }
  }
}

// ---------------------------------------------------------------------------
// Per-(b,y,x,head) attention over the Z column (Lq=1, Lk<=20, dh<=32).
// ---------------------------------------------------------------------------
__global__ void attn_k(const f16* __restrict__ kv, const f16* __restrict__ qp,
                       f16* __restrict__ ob,
                       int B, int Zz, int Yy, int Xx, int C, int dh) {
  long total = (long)B * Yy * Xx * 8;
  long i = (long)blockIdx.x * blockDim.x + threadIdx.x;
  if (i >= total) return;
  int h = (int)(i & 7); long col = i >> 3;
  int x = (int)(col % Xx); long t = col / Xx;
  int y = (int)(t % Yy);   int b = (int)(t / Yy);

  const f16* q = qp + ((size_t)y * Xx + x) * C + h * dh;
  float qv[32];
  for (int d = 0; d < dh; ++d) qv[d] = (float)q[d];

  const float scale = rsqrtf((float)dh);
  float sarr[32];
  float mx = -1e30f;
  for (int z = 0; z < Zz; ++z) {
    const f16* kr = kv + ((((size_t)b * Zz + z) * Yy + y) * Xx + x) * (size_t)(2 * C) + h * dh;
    float s = 0.f;
    for (int d = 0; d < dh; ++d) s += qv[d] * (float)kr[d];
    s *= scale;
    sarr[z] = s;
    mx = fmaxf(mx, s);
  }
  float ov[32];
  for (int d = 0; d < dh; ++d) ov[d] = 0.f;
  float sum = 0.f;
  for (int z = 0; z < Zz; ++z) {
    float w = __expf(sarr[z] - mx);
    sum += w;
    const f16* vr = kv + ((((size_t)b * Zz + z) * Yy + y) * Xx + x) * (size_t)(2 * C) + C + h * dh;
    for (int d = 0; d < dh; ++d) ov[d] += w * (float)vr[d];
  }
  float inv = 1.f / sum;
  f16* o = ob + (((size_t)b * Yy + y) * Xx + x) * (size_t)C + h * dh;
  for (int d = 0; d < dh; ++d) o[d] = (f16)(ov[d] * inv);
}

// ---------------------------------------------------------------------------
// Host orchestration
// ---------------------------------------------------------------------------
static inline long cdivl(long a, long b) { return (a + b - 1) / b; }

extern "C" void kernel_launch(void* const* d_in, const int* in_sizes, int n_in,
                              void* d_out, int out_size, void* d_ws, size_t ws_size,
                              hipStream_t stream) {
  (void)in_sizes; (void)n_in; (void)out_size; (void)ws_size;
  const int B = 4, Z = 20, Y = 80, X = 180, NPTS = 60000;
  const int OUTC = 112, OY = 80, OX = 180;

  const float* sp_feat = (const float*)d_in[0];
  const int*   sp_idx  = (const int*)d_in[1];
  const float* w_in    = (const float*)d_in[3];
  const float* b_in    = (const float*)d_in[4];

  struct LayerP {
    const float *w_sp,*b_sp,*g_sp,*be_sp,*m_sp,*v_sp;
    const float *w_a ,*b_a ,*g_a ,*be_a ,*m_a ,*v_a;
    const float *w_b ,*b_b ,*g_b ,*be_b ,*m_b ,*v_b;
    const float *q,*w_qkv,*b_qkv,*w_o,*b_o;
  } L[3];
  for (int l = 0; l < 3; ++l) {
    int o = 5 + l * 23;
    L[l].w_sp=(const float*)d_in[o+0];  L[l].b_sp=(const float*)d_in[o+1];
    L[l].g_sp=(const float*)d_in[o+2];  L[l].be_sp=(const float*)d_in[o+3];
    L[l].m_sp=(const float*)d_in[o+4];  L[l].v_sp=(const float*)d_in[o+5];
    L[l].w_a =(const float*)d_in[o+6];  L[l].b_a =(const float*)d_in[o+7];
    L[l].g_a =(const float*)d_in[o+8];  L[l].be_a=(const float*)d_in[o+9];
    L[l].m_a =(const float*)d_in[o+10]; L[l].v_a =(const float*)d_in[o+11];
    L[l].w_b =(const float*)d_in[o+12]; L[l].b_b =(const float*)d_in[o+13];
    L[l].g_b =(const float*)d_in[o+14]; L[l].be_b=(const float*)d_in[o+15];
    L[l].m_b =(const float*)d_in[o+16]; L[l].v_b =(const float*)d_in[o+17];
    L[l].q   =(const float*)d_in[o+18]; L[l].w_qkv=(const float*)d_in[o+19];
    L[l].b_qkv=(const float*)d_in[o+20];L[l].w_o =(const float*)d_in[o+21];
    L[l].b_o =(const float*)d_in[o+22];
  }

  const int Zl[3]  = {20, 10, 5},  Yl[3] = {80, 40, 20}, Xl[3] = {180, 90, 45};
  const int ZiL[3] = {20, 20, 10}, YiL[3]= {80, 80, 40}, XiL[3]= {180, 180, 90};
  const int Cl[3]  = {64, 128, 256}, CinL[3] = {64, 64, 128}, Sl[3] = {1, 2, 2};
  const int rL[3]  = {1, 2, 4}, choffL[3] = {0, 64, 96};
  const long NVo[3] = {(long)B*20*80*180, (long)B*10*40*90, (long)B*5*20*45};
  const long NV0 = NVo[0];

  char* ws = (char*)d_ws;
  size_t cur = 0;
  auto alloc = [&](size_t bytes) -> char* {
    cur = (cur + 255) & ~(size_t)255;
    char* p = ws + cur;
    cur += bytes;
    return p;
  };
  float* dense = (float*)alloc(NV0 * 4 * sizeof(float));
  float* cnt   = (float*)alloc(NV0 * sizeof(float));
  float* msk[3];
  for (int l = 0; l < 3; ++l) msk[l] = (float*)alloc(NVo[l] * sizeof(float));
  f16* zpad = (f16*)alloc(1024 * sizeof(f16));          // zeroed boundary pad
  const size_t actBytes = (size_t)NV0 * 64 * sizeof(f16);
  f16* slot0 = (f16*)alloc(actBytes);
  f16* slotA = (f16*)alloc(actBytes);
  f16* slotB = (f16*)alloc(actBytes);
  f16* kvbuf = (f16*)alloc((size_t)NV0 * 128 * sizeof(f16));
  f16* qf16  = (f16*)alloc((size_t)921600 * sizeof(f16));
  f16* qpbuf = (f16*)alloc((size_t)921600 * sizeof(f16));
  f16* obuf  = (f16*)alloc((size_t)3686400 * sizeof(f16));
  f16 *wsp[3], *wa[3], *wb[3], *wq[3], *wkv[3], *wo[3];
  float *scA[9], *shA[9];
  for (int l = 0; l < 3; ++l) {
    wsp[l] = (f16*)alloc((size_t)Cl[l] * CinL[l] * 27 * sizeof(f16));
    wa[l]  = (f16*)alloc((size_t)Cl[l] * Cl[l] * 27 * sizeof(f16));
    wb[l]  = (f16*)alloc((size_t)Cl[l] * Cl[l] * 27 * sizeof(f16));
    wq[l]  = (f16*)alloc((size_t)Cl[l] * Cl[l] * sizeof(f16));
    wkv[l] = (f16*)alloc((size_t)2 * Cl[l] * Cl[l] * sizeof(f16));
    wo[l]  = (f16*)alloc((size_t)Cl[l] * Cl[l] * sizeof(f16));
    for (int k = 0; k < 3; ++k) {
      scA[l*3+k] = (float*)alloc(Cl[l] * sizeof(float));
      shA[l*3+k] = (float*)alloc(Cl[l] * sizeof(float));
    }
  }

  const int TB = 256;

  // 1) zero dense grid + occupancy count + boundary pad, scatter points
  zero_f32<<<4096, TB, 0, stream>>>(dense, NV0 * 4);
  zero_f32<<<4096, TB, 0, stream>>>(cnt, NV0);
  zero_f32<<<1, TB, 0, stream>>>((float*)zpad, 512);
  scatter_pts<<<(unsigned)cdivl(NPTS, TB), TB, 0, stream>>>(sp_feat, sp_idx, dense, cnt, NPTS, Z, Y, X);

  // 2) input 1x1 conv (masked), f16 NDHWC activations
  conv_in_k<<<(unsigned)cdivl(NV0 * 64, TB), TB, 0, stream>>>(dense, cnt, w_in, b_in, slot0, NV0, 64);

  // 3) dilated masks per stage
  dilate_k<<<(unsigned)cdivl(NVo[0], TB), TB, 0, stream>>>(cnt,    msk[0], B, 20, 80, 180, 20, 80, 180, 1);
  dilate_k<<<(unsigned)cdivl(NVo[1], TB), TB, 0, stream>>>(msk[0], msk[1], B, 20, 80, 180, 10, 40, 90, 2);
  dilate_k<<<(unsigned)cdivl(NVo[2], TB), TB, 0, stream>>>(msk[1], msk[2], B, 10, 40, 90, 5, 20, 45, 2);

  // 4) fold BN+bias, pre-pack all weights into WMMA B-fragment layout
  for (int l = 0; l < 3; ++l) {
    int C = Cl[l];
    bn_prep<<<(unsigned)cdivl(C, TB), TB, 0, stream>>>(L[l].g_sp, L[l].be_sp, L[l].m_sp, L[l].v_sp, L[l].b_sp, scA[l*3+0], shA[l*3+0], C);
    bn_prep<<<(unsigned)cdivl(C, TB), TB, 0, stream>>>(L[l].g_a,  L[l].be_a,  L[l].m_a,  L[l].v_a,  L[l].b_a,  scA[l*3+1], shA[l*3+1], C);
    bn_prep<<<(unsigned)cdivl(C, TB), TB, 0, stream>>>(L[l].g_b,  L[l].be_b,  L[l].m_b,  L[l].v_b,  L[l].b_b,  scA[l*3+2], shA[l*3+2], C);
    long esp = (long)C * CinL[l] * 27, eaa = (long)C * C * 27;
    pack_conv_w<<<(unsigned)cdivl(esp, TB), TB, 0, stream>>>(L[l].w_sp, wsp[l], C, CinL[l]);
    pack_conv_w<<<(unsigned)cdivl(eaa, TB), TB, 0, stream>>>(L[l].w_a,  wa[l],  C, C);
    pack_conv_w<<<(unsigned)cdivl(eaa, TB), TB, 0, stream>>>(L[l].w_b,  wb[l],  C, C);
    long elin = (long)C * C;
    pack_lin_w<<<(unsigned)cdivl(elin, TB), TB, 0, stream>>>(L[l].w_qkv,          wq[l],  C,     C);
    pack_lin_w<<<(unsigned)cdivl(2*elin, TB), TB, 0, stream>>>(L[l].w_qkv + elin, wkv[l], 2 * C, C);
    pack_lin_w<<<(unsigned)cdivl(elin, TB), TB, 0, stream>>>(L[l].w_o,            wo[l],  C,     C);
  }

  // 5) backbone stages
  f16* curAct = slot0;
  for (int l = 0; l < 3; ++l) {
    const int C = Cl[l], Cin = CinL[l], s = Sl[l];
    const int Zi = ZiL[l], Yi = YiL[l], Xi = XiL[l];
    const int Zo = Zl[l], Yo = Yl[l], Xo = Xl[l];
    const int ntX = (Xo + 15) >> 4;
    const long Stiles = (long)B * Zo * Yo * ntX;

    const int KCsp = Cin >> 5, KCaa = C >> 5;
    const unsigned cblk = (unsigned)((long)(C >> 6) * cdivl(Stiles, 8));
    const size_t sh_sp = (size_t)KCsp * 8192;   // 2 * 4*KC KB
    const size_t sh_aa = (size_t)KCaa * 8192;

    conv3d_wmma<<<cblk, TB, sh_sp, stream>>>(curAct, slotB, wsp[l], zpad,
                                             scA[l*3+0], shA[l*3+0], msk[l],
                                             B, Zi, Yi, Xi, Cin, Zo, Yo, Xo, C, s);
    conv3d_wmma<<<cblk, TB, sh_aa, stream>>>(slotB, slot0, wa[l], zpad,
                                             scA[l*3+1], shA[l*3+1], msk[l],
                                             B, Zo, Yo, Xo, C, Zo, Yo, Xo, C, 1);
    conv3d_wmma<<<cblk, TB, sh_aa, stream>>>(slot0, slotA, wb[l], zpad,
                                             scA[l*3+2], shA[l*3+2], msk[l],
                                             B, Zo, Yo, Xo, C, Zo, Yo, Xo, C, 1);

    // K/V projection over all voxels: [NVo, C] @ [C, 2C] -> kvbuf
    const long Mkv = NVo[l];
    {
      long blocks = (long)((2 * C) >> 6) * cdivl(cdivl(Mkv, 16), 8);
      gemm_wmma<<<(unsigned)blocks, TB, (size_t)KCaa * 4096, stream>>>(
          slotA, wkv[l], L[l].b_qkv + C, zpad, (int)Mkv, C, 2 * C,
          0, kvbuf, nullptr, 0, 0, 0, 0, 0, 0, 0);
    }
    // Q projection (per-(y,x), broadcast over batch)
    const int yy = Yo, xx = Xo;
    const long Mq = (long)yy * xx;
    f32_to_f16_k<<<(unsigned)cdivl(Mq * C, TB), TB, 0, stream>>>(L[l].q, qf16, Mq * C);
    {
      long blocks = (long)(C >> 6) * cdivl(cdivl(Mq, 16), 8);
      gemm_wmma<<<(unsigned)blocks, TB, (size_t)KCaa * 4096, stream>>>(
          qf16, wq[l], L[l].b_qkv, zpad, (int)Mq, C, C,
          0, qpbuf, nullptr, 0, 0, 0, 0, 0, 0, 0);
    }
    // column attention
    const int dh = C / 8;
    attn_k<<<(unsigned)cdivl((long)B * yy * xx * 8, TB), TB, 0, stream>>>(
        kvbuf, qpbuf, obuf, B, Zo, yy, xx, C, dh);
    // output projection with fused BEV pixel-shuffle scatter into d_out
    const long Mo = (long)B * yy * xx;
    {
      long blocks = (long)(C >> 6) * cdivl(cdivl(Mo, 16), 8);
      gemm_wmma<<<(unsigned)blocks, TB, (size_t)KCaa * 4096, stream>>>(
          obuf, wo[l], L[l].b_o, zpad, (int)Mo, C, C,
          1, nullptr, (float*)d_out, yy, xx, rL[l], choffL[l], OUTC, OY, OX);
    }
    curAct = slotA;
  }
}
